// MonotonicConvolutionalMultiheadAttention_70566312673704
// MI455X (gfx1250) — matrix operand order repaired and checked by reading, not compiled, not run on hardware
//
#include <hip/hip_runtime.h>
#include <hip/hip_bf16.h>

#define HEADS 6
#define DHEAD 64
#define ALLH 384
#define KS 9
#define PADW 4
#define B_ 8
#define S_ 1024
#define H_ 768
#define MS_ (B_ * S_)   // 8192 token rows

typedef __attribute__((ext_vector_type(16))) __bf16 v16bf;
typedef __attribute__((ext_vector_type(8)))  float  v8f;
typedef __attribute__((ext_vector_type(4)))  unsigned int v4u;
typedef __attribute__((ext_vector_type(8)))  int v8i;
typedef __attribute__((ext_vector_type(4)))  int v4i;

__device__ __forceinline__ __bf16 tobf(float x) { return (__bf16)x; }

__device__ __forceinline__ void cvt4(v16bf& a, int base, float4 x) {
  a[base] = tobf(x.x); a[base + 1] = tobf(x.y);
  a[base + 2] = tobf(x.z); a[base + 3] = tobf(x.w);
}
__device__ __forceinline__ float4 mul4(float4 a, float4 b) {
  return make_float4(a.x * b.x, a.y * b.y, a.z * b.z, a.w * b.w);
}

// ---------------------------------------------------------------------------
// GEMM: C[M,N] = (A .* A2?) [M,K] @ W (strided) + bias?
// bf16 WMMA 16x16x32, fp32 accumulate. 128 threads = 4 waves; block covers
// 64 rows x 16 cols. W tile (32K x 16N) is cooperatively staged to LDS as
// bf16, transposed to [n][k] so each lane's B fragment is one 32B LDS read.
// OUT_MODE: 0 = fp32 row-major, 1 = bf16 row-major, 2 = bf16 transposed [N][M]
// ---------------------------------------------------------------------------
template<bool HAS_A2, int OUT_MODE>
__global__ __launch_bounds__(128)
void gemm_bf16_wmma(const float* __restrict__ A, const float* __restrict__ A2,
                    const float* __restrict__ W, const float* __restrict__ bias,
                    void* __restrict__ Cout, int M, int N, int K, int wSK, int wSN)
{
  __shared__ __bf16 Wt[2][16][32];   // [buf][n][k], 2 KB
  const int tid  = threadIdx.x;
  const int lane = tid & 31;
  const int wave = tid >> 5;
  const int half = lane >> 4;        // 0: lanes 0-15, 1: lanes 16-31
  const int lp   = lane & 15;
  const int m0   = blockIdx.x * 64 + wave * 16;
  const int n0   = blockIdx.y * 16;
  const int mrow = m0 + lp;
  const int ncol = n0 + lp;

  const float* arow  = A + (long)mrow * K;
  const float* a2row = HAS_A2 ? (A2 + (long)mrow * K) : nullptr;

  // staging: thread t covers W row kr = t>>2, cols snc..snc+3
  const int skr = tid >> 2;
  const int snc = (tid & 3) * 4;
  auto stage = [&](int buf, int kk) {
    #pragma unroll
    for (int i = 0; i < 4; ++i) {
      int n = n0 + snc + i;
      float wv = (n < N) ? W[(long)(kk + skr) * wSK + (long)n * wSN] : 0.f;
      Wt[buf][snc + i][skr] = tobf(wv);
    }
  };

  stage(0, 0);
  v8f acc = {};
  int buf = 0;
  for (int kk = 0; kk < K; kk += 32) {
    __syncthreads();
    if (kk + 32 < K) stage(buf ^ 1, kk + 32);

    v16bf a, bfrag;
    float4 x0 = *(const float4*)(arow + kk + half * 16);
    float4 x1 = *(const float4*)(arow + kk + half * 16 + 4);
    float4 y0 = *(const float4*)(arow + kk + half * 16 + 8);
    float4 y1 = *(const float4*)(arow + kk + half * 16 + 12);
    if (HAS_A2) {
      x0 = mul4(x0, *(const float4*)(a2row + kk + half * 16));
      x1 = mul4(x1, *(const float4*)(a2row + kk + half * 16 + 4));
      y0 = mul4(y0, *(const float4*)(a2row + kk + half * 16 + 8));
      y1 = mul4(y1, *(const float4*)(a2row + kk + half * 16 + 12));
    }
    cvt4(a, 0, x0); cvt4(a, 4, x1); cvt4(a, 8, y0); cvt4(a, 12, y1);
    bfrag = *(const v16bf*)&Wt[buf][lp][half * 16];

    acc = __builtin_amdgcn_wmma_f32_16x16x32_bf16(false, a, false, bfrag,
                                                  (short)0, acc, false, false);
    buf ^= 1;
  }

  if (OUT_MODE == 0) {
    if (ncol < N) {
      float bv = bias ? bias[ncol] : 0.f;
      float* C = (float*)Cout;
      #pragma unroll
      for (int vv = 0; vv < 8; ++vv)
        C[(long)(m0 + vv + half * 8) * N + ncol] = acc[vv] + bv;
    }
  } else if (OUT_MODE == 1) {
    if (ncol < N) {
      float bv = bias ? bias[ncol] : 0.f;
      __bf16* C = (__bf16*)Cout;
      #pragma unroll
      for (int vv = 0; vv < 8; ++vv)
        C[(long)(m0 + vv + half * 8) * N + ncol] = tobf(acc[vv] + bv);
    }
  } else {
    if (ncol < N) {
      __bf16* C = (__bf16*)Cout;
      union { __bf16 h[8]; uint4 u; } p;
      #pragma unroll
      for (int vv = 0; vv < 8; ++vv) p.h[vv] = tobf(acc[vv]);
      *(uint4*)(C + (long)ncol * M + m0 + half * 8) = p.u;   // 16B store
    }
  }
}

// ---------------------------------------------------------------------------
// Depthwise 'same' conv: xconv[b,s,c] = sum_t K[b,s+t-4,c]*dw[c,t]
// ---------------------------------------------------------------------------
__global__ __launch_bounds__(256)
void dwconv_kernel(const float* __restrict__ Kin, const float* __restrict__ dw,
                   float* __restrict__ xconv)
{
  long idx = (long)blockIdx.x * blockDim.x + threadIdx.x;
  if (idx >= (long)MS_ * H_) return;
  int  c  = (int)(idx % H_);
  long bs = idx / H_;
  int  s  = (int)(bs % S_);
  float accv = 0.f;
  #pragma unroll
  for (int t = 0; t < KS; ++t) {
    int sj = s + t - PADW;
    if (sj >= 0 && sj < S_)
      accv += Kin[(bs - s + sj) * H_ + c] * dw[c * KS + t];
  }
  xconv[idx] = accv;
}

// ---------------------------------------------------------------------------
// Span-dynamic conv: softmax over 9 taps, mix co windows -> out[..., 384:768]
// ---------------------------------------------------------------------------
__global__ __launch_bounds__(256)
void convout_kernel(const float* __restrict__ kernlog, const float* __restrict__ co,
                    float* __restrict__ out)
{
  long idx = (long)blockIdx.x * blockDim.x + threadIdx.x;
  if (idx >= (long)MS_ * ALLH) return;
  int  hd = (int)(idx % ALLH);
  long bs = idx / ALLH;
  int  h  = hd / DHEAD;
  int  s  = (int)(bs % S_);

  const float* kl = kernlog + bs * (HEADS * KS) + h * KS;
  float mx = kl[0];
  #pragma unroll
  for (int t = 1; t < KS; ++t) mx = fmaxf(mx, kl[t]);
  float w[KS]; float sum = 0.f;
  #pragma unroll
  for (int t = 0; t < KS; ++t) { w[t] = __expf(kl[t] - mx); sum += w[t]; }
  float inv = 1.f / sum;

  float accv = 0.f;
  #pragma unroll
  for (int t = 0; t < KS; ++t) {
    int sj = s + t - PADW;
    if (sj >= 0 && sj < S_)
      accv += w[t] * inv * co[(bs - s + sj) * ALLH + hd];
  }
  out[bs * (2 * ALLH) + ALLH + hd] = accv;
}

// ---------------------------------------------------------------------------
// Attention with monotonic reweighting. Block = (16-row i-tile, head, batch),
// 512 threads = 16 waves. q tile staged via TDM tensor_load_to_lds; k is bf16
// row-major; v is bf16 transposed [ALLH][M] -> single v16bf B-fragment loads.
// ---------------------------------------------------------------------------
__global__ __launch_bounds__(512)
void attn_kernel(const float* __restrict__ q, const __bf16* __restrict__ k16,
                 const __bf16* __restrict__ vT, const float* __restrict__ gammas,
                 const int* __restrict__ mask, float* __restrict__ out)
{
  __shared__ float sc[16 * S_];      // 64 KB score/prob tile
  __shared__ float qtile[16 * 64];   // 4 KB q tile
  const int i0   = blockIdx.x * 16;
  const int h    = blockIdx.y;
  const int b    = blockIdx.z;
  const int lane = threadIdx.x & 31;
  const int wave = threadIdx.x >> 5;
  const int half = lane >> 4;
  const int lp   = lane & 15;

  // ---- Stage q tile [16 x 64] into LDS via TDM (6-arg toolchain form) ----
#if __has_builtin(__builtin_amdgcn_tensor_load_to_lds) && __has_builtin(__builtin_amdgcn_s_wait_tensorcnt)
  if (wave == 0) {
    unsigned long long ga =
        (unsigned long long)(uintptr_t)(q + ((long)b * S_ + i0) * ALLH + h * DHEAD);
    unsigned ldsoff = (unsigned)(uintptr_t)(void*)qtile;
    v4u g0 = { 1u,                                  // count=1 valid descriptor
               ldsoff,                               // lds_addr
               (unsigned)(ga & 0xffffffffu),         // global_addr[31:0]
               (unsigned)((ga >> 32) & 0x01ffffffu) | (2u << 30) }; // addr[56:32] | type=2
    v8i g1 = { (int)(2u << 16),                      // data_size=2 (4B), no flags
               (int)((DHEAD & 0xffffu) << 16),        // tensor_dim0 lo16 = 64
               (int)((unsigned)S_ << 16),             // dim0 hi=0 | tensor_dim1 lo16 = 1024
               (int)((unsigned)DHEAD << 16),          // dim1 hi=0 | tile_dim0 = 64
               16,                                    // tile_dim1=16 | tile_dim2=0
               ALLH,                                  // tensor_dim0_stride lo32 = 384
               0, 0 };
    v4i g2 = { 0, 0, 0, 0 };
    v4i g3 = { 0, 0, 0, 0 };
    v8i g4 = { 0, 0, 0, 0, 0, 0, 0, 0 };
    __builtin_amdgcn_tensor_load_to_lds(g0, g1, g2, g3, g4, 0);
    __builtin_amdgcn_s_wait_tensorcnt(0);
  }
#else
  for (int e = threadIdx.x; e < 16 * 64; e += 512) {
    int r = e >> 6, d = e & 63;
    qtile[e] = q[((long)b * S_ + i0 + r) * ALLH + h * DHEAD + d];
  }
#endif
  __syncthreads();

  const __bf16* kb = k16 + (long)b * S_ * ALLH + h * DHEAD;
  const __bf16* vb = vT + (long)(h * DHEAD) * MS_ + (long)b * S_;

  // ---- Phase 1: scores[i,j] = (q_i . k_j) / 8 ----
  for (int jj = 0; jj < 4; ++jj) {
    int j0 = wave * 64 + jj * 16;
    v8f acc = {};
    const float* qrow = qtile + lp * 64;
    const __bf16* krow = kb + (long)(j0 + lp) * ALLH;
    #pragma unroll
    for (int kk = 0; kk < DHEAD; kk += 32) {
      v16bf a, bfrag;
      cvt4(a, 0,  *(const float4*)(qrow + kk + half * 16));
      cvt4(a, 4,  *(const float4*)(qrow + kk + half * 16 + 4));
      cvt4(a, 8,  *(const float4*)(qrow + kk + half * 16 + 8));
      cvt4(a, 12, *(const float4*)(qrow + kk + half * 16 + 12));
      bfrag = *(const v16bf*)(krow + kk + half * 16);   // 16 contiguous bf16
      acc = __builtin_amdgcn_wmma_f32_16x16x32_bf16(false, a, false, bfrag,
                                                    (short)0, acc, false, false);
    }
    #pragma unroll
    for (int vv = 0; vv < 8; ++vv)
      sc[(vv + half * 8) * S_ + j0 + lp] = acc[vv] * 0.125f;
  }
  __syncthreads();

  // ---- Phase 2: per-row monotonic reweight + second softmax ----
  {
    const int r = wave, irow = i0 + r;
    float* row = sc + r * S_;
    const int* mrow = mask + (long)b * S_;
    float g  = gammas[h];
    float gamma = -((g > 20.f) ? g : log1pf(__expf(g)));

    float mx = -3.0e38f;
    for (int t = 0; t < 32; ++t) {
      int j = lane * 32 + t;
      mx = fmaxf(mx, (mrow[j] != 0) ? row[j] : -3.0e38f);
    }
    #pragma unroll
    for (int off = 16; off > 0; off >>= 1) mx = fmaxf(mx, __shfl_xor(mx, off, 32));

    float se = 0.f;
    for (int t = 0; t < 32; ++t) {
      int j = lane * 32 + t;
      if (mrow[j] != 0) se += __expf(row[j] - mx);
    }
    #pragma unroll
    for (int off = 16; off > 0; off >>= 1) se += __shfl_xor(se, off, 32);
    float invSE = (se > 0.f) ? 1.f / se : 0.f;

    float lsum = 0.f;
    for (int t = 0; t < 32; ++t) {
      int j = lane * 32 + t;
      if (mrow[j] != 0) lsum += __expf(row[j] - mx) * invSE;
    }
    float pre = lsum;
    #pragma unroll
    for (int off = 1; off < 32; off <<= 1) {
      float o = __shfl_up(pre, off, 32);
      if (lane >= off) pre += o;
    }
    float tot = __shfl(pre, 31, 32);
    float run = pre - lsum;   // exclusive prefix

    for (int t = 0; t < 32; ++t) {
      int j = lane * 32 + t;
      bool mm = (mrow[j] != 0);
      float p = mm ? __expf(row[j] - mx) * invSE : 0.f;
      run += p;
      float rem  = tot - run;
      float pos  = fabsf((float)(irow - j));
      float dist = sqrtf(fmaxf(rem * pos, 0.f));
      float eff  = fminf(fmaxf(__expf(dist * gamma), 1e-5f), 1e5f);
      row[j] = mm ? row[j] * eff : -1e8f;
    }

    float mx2 = -3.0e38f;
    for (int t = 0; t < 32; ++t) mx2 = fmaxf(mx2, row[lane * 32 + t]);
    #pragma unroll
    for (int off = 16; off > 0; off >>= 1) mx2 = fmaxf(mx2, __shfl_xor(mx2, off, 32));
    float s2 = 0.f;
    for (int t = 0; t < 32; ++t) s2 += __expf(row[lane * 32 + t] - mx2);
    #pragma unroll
    for (int off = 16; off > 0; off >>= 1) s2 += __shfl_xor(s2, off, 32);
    float inv2 = 1.f / s2;
    for (int t = 0; t < 32; ++t) {
      int j = lane * 32 + t;
      row[j] = __expf(row[j] - mx2) * inv2;
    }
  }
  __syncthreads();

  // ---- Phase 3: ctx[16 x 64] = probs @ v  (v transposed bf16: 1-load frags) ----
  if (wave < 4) {
    const int nt = wave;
    v8f acc = {};
    const float* arow = sc + lp * S_;
    const __bf16* vrow = vb + (long)(nt * 16 + lp) * MS_;
    for (int kk = 0; kk < S_; kk += 32) {
      v16bf a, bfrag;
      cvt4(a, 0,  *(const float4*)(arow + kk + half * 16));
      cvt4(a, 4,  *(const float4*)(arow + kk + half * 16 + 4));
      cvt4(a, 8,  *(const float4*)(arow + kk + half * 16 + 8));
      cvt4(a, 12, *(const float4*)(arow + kk + half * 16 + 12));
      bfrag = *(const v16bf*)(vrow + kk + half * 16);   // 16 contiguous bf16
      acc = __builtin_amdgcn_wmma_f32_16x16x32_bf16(false, a, false, bfrag,
                                                    (short)0, acc, false, false);
    }
    #pragma unroll
    for (int vv = 0; vv < 8; ++vv)
      out[((long)b * S_ + i0 + vv + half * 8) * (2 * ALLH) + h * DHEAD + nt * 16 + lp]
          = acc[vv];
  }
}

// ---------------------------------------------------------------------------
extern "C" void kernel_launch(void* const* d_in, const int* in_sizes, int n_in,
                              void* d_out, int out_size, void* d_ws, size_t ws_size,
                              hipStream_t stream) {
  const float* Q        = (const float*)d_in[0];
  const float* Kin      = (const float*)d_in[1];
  const float* V        = (const float*)d_in[2];
  const float* Wq       = (const float*)d_in[3];
  const float* Wk       = (const float*)d_in[4];
  const float* Wv       = (const float*)d_in[5];
  const float* dw       = (const float*)d_in[6];
  const float* pw       = (const float*)d_in[7];
  const float* sep_bias = (const float*)d_in[8];
  const float* Wck      = (const float*)d_in[9];
  const float* bck      = (const float*)d_in[10];
  const float* Wco      = (const float*)d_in[11];
  const float* bco      = (const float*)d_in[12];
  const float* gammas   = (const float*)d_in[13];
  const int*   mask     = (const int*)d_in[14];
  float* out = (float*)d_out;

  const long MS = MS_;
  char* w = (char*)d_ws;
  float*  ws_q  = (float*)w;  w += (long)MS * ALLH * 4;
  float*  ws_co = (float*)w;  w += (long)MS * ALLH * 4;
  float*  ws_kc = (float*)w;  w += (long)MS * ALLH * 4;
  float*  ws_x  = (float*)w;  w += (long)MS * H_ * 4;
  float*  ws_kl = (float*)w;  w += (long)MS * HEADS * KS * 4;
  __bf16* ws_k  = (__bf16*)w; w += (long)MS * ALLH * 2;
  __bf16* ws_vT = (__bf16*)w; w += (long)MS * ALLH * 2;

  dim3 g1((unsigned)(MS / 64), ALLH / 16);
  gemm_bf16_wmma<false, 0><<<g1, 128, 0, stream>>>(Q,   nullptr, Wq,  nullptr, ws_q,  (int)MS, ALLH, H_, ALLH, 1);
  gemm_bf16_wmma<false, 1><<<g1, 128, 0, stream>>>(Kin, nullptr, Wk,  nullptr, ws_k,  (int)MS, ALLH, H_, ALLH, 1);
  gemm_bf16_wmma<false, 2><<<g1, 128, 0, stream>>>(V,   nullptr, Wv,  nullptr, ws_vT, (int)MS, ALLH, H_, ALLH, 1);
  gemm_bf16_wmma<false, 0><<<g1, 128, 0, stream>>>(V,   nullptr, Wco, bco,     ws_co, (int)MS, ALLH, H_, ALLH, 1);

  long nx = MS * H_;
  dwconv_kernel<<<(unsigned)((nx + 255) / 256), 256, 0, stream>>>(Kin, dw, ws_x);
  // keyconv = xconv @ pw^T + sep_bias  (pw is [ALLH,H] row-major -> wSK=1, wSN=H)
  gemm_bf16_wmma<false, 0><<<g1, 128, 0, stream>>>(ws_x, nullptr, pw, sep_bias, ws_kc, (int)MS, ALLH, H_, 1, H_);
  // kern logits = (keyconv .* q) @ Wck + bck, N=54 (grid padded to 64 cols)
  dim3 g2((unsigned)(MS / 64), (HEADS * KS + 15) / 16);
  gemm_bf16_wmma<true, 0><<<g2, 128, 0, stream>>>(ws_kc, ws_q, Wck, bck, ws_kl, (int)MS, HEADS * KS, ALLH, HEADS * KS, 1);

  long nc = MS * ALLH;
  convout_kernel<<<(unsigned)((nc + 255) / 256), 256, 0, stream>>>(ws_kl, ws_co, out);

  dim3 ga(S_ / 16, HEADS, B_);
  attn_kernel<<<ga, 512, 0, stream>>>(ws_q, ws_k, ws_vT, gammas, mask, out);
}
